// ContextHypergraphAttention_20109036880615
// MI455X (gfx1250) — matrix operand order repaired and checked by reading, not compiled, not run on hardware
//
#include <hip/hip_runtime.h>
#include <hip/hip_bf16.h>

typedef __attribute__((ext_vector_type(16))) _Float16 v16h;
typedef __attribute__((ext_vector_type(8)))  _Float16 v8h;
typedef __attribute__((ext_vector_type(8)))  float    v8f;
typedef __attribute__((ext_vector_type(4)))  unsigned int v4u;
typedef __attribute__((ext_vector_type(8)))  int      v8i;
typedef __attribute__((ext_vector_type(4)))  int      v4i;

#define B_ 4
#define N_ 4096
#define E_ 128
#define KT 64          // keys per tile
#define NT (N_ / KT)   // key tiles

// log2(e) / sqrt(E): folded into Q so softmax runs in the exp2 domain.
#define QSCALE (1.4426950408889634f * 0.08838834764831845f)

// ---------------------------------------------------------------------------
// TDM: issue a 2D tensor_load_to_lds.  Descriptor layout per CDNA5 ISA §8.
//   group0: count=1 | lds_addr | global_addr(57b) | type=2
//   group1: data_size=2B, tensor_dim0/1, tile_dim0/1, tensor_dim0_stride
// ---------------------------------------------------------------------------
__device__ __forceinline__ void tdm_load_2d(unsigned int lds_byte_addr,
                                            const void* gaddr,
                                            unsigned int tile_d0,
                                            unsigned int tile_d1,
                                            unsigned int tensor_d0,
                                            unsigned int tensor_d1,
                                            unsigned int stride0_elems)
{
    unsigned long long ga = (unsigned long long)(uintptr_t)gaddr;
    v4u g0;
    g0[0] = 1u;                                              // count=1, user mode
    g0[1] = lds_byte_addr;                                   // lds_addr
    g0[2] = (unsigned int)(ga & 0xFFFFFFFFu);                // global_addr[31:0]
    g0[3] = (unsigned int)((ga >> 32) & 0x1FFFFFFu)          // global_addr[56:32]
          | (2u << 30);                                      // type=2 (image)
    v8i g1;
    g1[0] = (int)(1u << 16);                                 // data_size=1 -> 2 bytes
    g1[1] = (int)((tensor_d0 & 0xFFFFu) << 16);              // tensor_dim0[15:0]
    g1[2] = (int)(((tensor_d0 >> 16) & 0xFFFFu)
          | ((tensor_d1 & 0xFFFFu) << 16));                  // dim0 hi | dim1 lo
    g1[3] = (int)(((tensor_d1 >> 16) & 0xFFFFu)
          | ((tile_d0 & 0xFFFFu) << 16));                    // dim1 hi | tile_dim0
    g1[4] = (int)(tile_d1 & 0xFFFFu);                        // tile_dim1 (tile_dim2=0)
    g1[5] = (int)stride0_elems;                              // tensor_dim0_stride lo
    g1[6] = 0;                                               // stride0 hi | stride1 lo
    g1[7] = 0;
    v4i z4 = {0, 0, 0, 0};
#if __clang_major__ >= 23
    v8i z8 = {0, 0, 0, 0, 0, 0, 0, 0};
    __builtin_amdgcn_tensor_load_to_lds(g0, g1, z4, z4, z8, 0);
#else
    __builtin_amdgcn_tensor_load_to_lds(g0, g1, z4, z4, 0);
#endif
}

// ---------------------------------------------------------------------------
// Stage 1: Q/K/V projection.  Q[m,f] = sum_e X[m,e]*W[f,e] + b[f]
// One wave computes one 16x16 tile.  Q is pre-scaled by QSCALE.
// Q,K row-major f16 [B*N][E]; V transposed f16 [B][E][N].
// ---------------------------------------------------------------------------
__global__ void __launch_bounds__(256)
qkv_proj_kernel(const float* __restrict__ X,
                const float* __restrict__ Wq, const float* __restrict__ bq,
                const float* __restrict__ Wk, const float* __restrict__ bk,
                const float* __restrict__ Wv, const float* __restrict__ bv,
                _Float16* __restrict__ Qh, _Float16* __restrict__ Kh,
                _Float16* __restrict__ Vt)
{
    const int lane = threadIdx.x & 31;
    const int wave = threadIdx.x >> 5;
    const int lr   = lane & 15;
    const int hi   = lane >> 4;
    const int m0   = blockIdx.x * 16;
    const int n0   = wave * 16;
    const int mat  = blockIdx.y;                // 0:Q 1:K 2:V

    const float* W;  const float* bias;
    if      (mat == 0) { W = Wq; bias = bq; }
    else if (mat == 1) { W = Wk; bias = bk; }
    else               { W = Wv; bias = bv; }

    v8f acc = {};
    #pragma unroll
    for (int kk = 0; kk < 4; ++kk) {
        const int k0 = kk * 32;
        const float* xrow = X + (size_t)(m0 + lr) * E_ + k0 + 16 * hi;
        v16h a;
        #pragma unroll
        for (int i = 0; i < 8; ++i) { a[i] = (_Float16)xrow[i]; a[8 + i] = (_Float16)xrow[8 + i]; }
        const float* wrow = W + (size_t)(n0 + lr) * E_ + k0 + 16 * hi;
        v16h bfrag;
        #pragma unroll
        for (int i = 0; i < 16; ++i) bfrag[i] = (_Float16)wrow[i];
        acc = __builtin_amdgcn_wmma_f32_16x16x32_f16(false, a, false, bfrag,
                                                     (short)0, acc, false, false);
    }

    const float bb = bias[n0 + lr];
    if (mat == 0) {
        #pragma unroll
        for (int r = 0; r < 8; ++r) {
            const int m = m0 + r + 8 * hi;
            Qh[(size_t)m * E_ + n0 + lr] = (_Float16)((acc[r] + bb) * QSCALE);
        }
    } else if (mat == 1) {
        #pragma unroll
        for (int r = 0; r < 8; ++r) {
            const int m = m0 + r + 8 * hi;
            Kh[(size_t)m * E_ + n0 + lr] = (_Float16)(acc[r] + bb);
        }
    } else {
        #pragma unroll
        for (int r = 0; r < 8; ++r) {
            const int m  = m0 + r + 8 * hi;     // global row = b*N_ + n
            const int bi = m / N_;
            const int n  = m - bi * N_;
            Vt[((size_t)bi * E_ + (n0 + lr)) * N_ + n] = (_Float16)(acc[r] + bb);
        }
    }
}

// ---------------------------------------------------------------------------
// Stage 2: flash attention.  8 waves/block = 128 query rows; KT=64 key tiles
// TDM double-buffers K and V tiles into LDS; compute reads LDS fragments.
// The context bias is softmax-invariant (constant along the key axis) -> omitted.
// ---------------------------------------------------------------------------
__global__ void __launch_bounds__(256)
attn_kernel(const _Float16* __restrict__ Qh, const _Float16* __restrict__ Kh,
            const _Float16* __restrict__ Vt, float* __restrict__ out)
{
    constexpr int PSTR = 72;
    __shared__ _Float16 Ks[2][KT * E_];      // [key][e]  16 KB each
    __shared__ _Float16 Vs[2][E_ * KT];      // [e][key]  16 KB each
    __shared__ _Float16 ldsP[8 * 16 * PSTR]; // per-wave P bounce

    const int lane = threadIdx.x & 31;
    const int wave = threadIdx.x >> 5;
    const int lr   = lane & 15;
    const int hi   = lane >> 4;
    const int q0   = blockIdx.x * 128 + wave * 16;   // global query row over B*N
    const int b    = q0 / N_;
    _Float16* P    = ldsP + wave * 16 * PSTR;

    const _Float16* Kbase = Kh + (size_t)b * N_ * E_;
    const _Float16* Vbase = Vt + (size_t)b * E_ * N_;

    // Preload Q A-fragments (already scaled by log2(e)/sqrt(E))
    v16h qf[4];
    const _Float16* qrow = Qh + (size_t)(q0 + lr) * E_;
    #pragma unroll
    for (int kk = 0; kk < 4; ++kk) {
        const int k0 = kk * 32;
        v8h lo = *(const v8h*)(qrow + k0 + 16 * hi);
        v8h hu = *(const v8h*)(qrow + k0 + 8 + 16 * hi);
        #pragma unroll
        for (int i = 0; i < 8; ++i) { qf[kk][i] = lo[i]; qf[kk][8 + i] = hu[i]; }
    }

    // Prologue: TDM tile 0 into buffer 0
    if (wave == 0) {
        tdm_load_2d((unsigned int)(uintptr_t)&Ks[0][0], Kbase,
                    /*tile*/ E_, KT, /*tensor*/ E_, KT, /*stride0*/ E_);
        tdm_load_2d((unsigned int)(uintptr_t)&Vs[0][0], Vbase,
                    /*tile*/ KT, E_, /*tensor*/ KT, E_, /*stride0*/ N_);
        __builtin_amdgcn_s_wait_tensorcnt(0);
    }
    __syncthreads();

    v8f O[8] = {};
    float mstate[8], lstate[8];
    #pragma unroll
    for (int r = 0; r < 8; ++r) { mstate[r] = -1e30f; lstate[r] = 0.0f; }

    for (int it = 0; it < NT; ++it) {
        const int cur = it & 1;
        const int nxt = cur ^ 1;
        if (wave == 0 && it + 1 < NT) {
            const int kb1 = (it + 1) * KT;
            tdm_load_2d((unsigned int)(uintptr_t)&Ks[nxt][0],
                        Kbase + (size_t)kb1 * E_,
                        E_, KT, E_, KT, E_);
            tdm_load_2d((unsigned int)(uintptr_t)&Vs[nxt][0],
                        Vbase + kb1,
                        KT, E_, KT, E_, N_);
        }

        // ---- S = Q K^T (log2 domain): 16 WMMAs from LDS K tile
        v8f S[4];
        #pragma unroll
        for (int t = 0; t < 4; ++t) {
            v8f s = {};
            const _Float16* krow = &Ks[cur][(t * 16 + lr) * E_ + 16 * hi];
            #pragma unroll
            for (int kk = 0; kk < 4; ++kk) {
                v16h bf = *(const v16h*)(krow + kk * 32);
                s = __builtin_amdgcn_wmma_f32_16x16x32_f16(false, qf[kk], false, bf,
                                                           (short)0, s, false, false);
            }
            S[t] = s;
        }

        // ---- online softmax (exp2 domain); row m=r+8*hi lives in a 16-lane group
        float alpha[8];
        #pragma unroll
        for (int r = 0; r < 8; ++r) {
            float v = fmaxf(fmaxf(S[0][r], S[1][r]), fmaxf(S[2][r], S[3][r]));
            #pragma unroll
            for (int off = 1; off <= 8; off <<= 1) v = fmaxf(v, __shfl_xor(v, off, 32));
            const float mnew = fmaxf(mstate[r], v);
            alpha[r]  = __builtin_amdgcn_exp2f(mstate[r] - mnew);
            mstate[r] = mnew;
            float rs = 0.0f;
            #pragma unroll
            for (int t = 0; t < 4; ++t) {
                const float p = __builtin_amdgcn_exp2f(S[t][r] - mnew);
                S[t][r] = p;
                rs += p;
            }
            #pragma unroll
            for (int off = 1; off <= 8; off <<= 1) rs += __shfl_xor(rs, off, 32);
            lstate[r] = lstate[r] * alpha[r] + rs;
        }
        #pragma unroll
        for (int j = 0; j < 8; ++j)
            #pragma unroll
            for (int r = 0; r < 8; ++r) O[j][r] *= alpha[r];

        // ---- bounce P through LDS: C-layout -> row-major P[16][PSTR]
        #pragma unroll
        for (int t = 0; t < 4; ++t)
            #pragma unroll
            for (int r = 0; r < 8; ++r)
                P[(r + 8 * hi) * PSTR + t * 16 + lr] = (_Float16)S[t][r];

        // ---- O += P * V : 16 WMMAs from LDS V tile
        #pragma unroll
        for (int kk = 0; kk < 2; ++kk) {
            const int k0 = kk * 32;
            const _Float16* prow = P + lr * PSTR + k0 + 16 * hi;
            v8h lo = *(const v8h*)(prow);
            v8h hu = *(const v8h*)(prow + 8);
            v16h ap;
            #pragma unroll
            for (int i = 0; i < 8; ++i) { ap[i] = lo[i]; ap[8 + i] = hu[i]; }
            #pragma unroll
            for (int j = 0; j < 8; ++j) {
                const _Float16* vrow = &Vs[cur][(j * 16 + lr) * KT + k0 + 16 * hi];
                v16h bf = *(const v16h*)vrow;
                O[j] = __builtin_amdgcn_wmma_f32_16x16x32_f16(false, ap, false, bf,
                                                              (short)0, O[j], false, false);
            }
        }

        if (wave == 0 && it + 1 < NT) __builtin_amdgcn_s_wait_tensorcnt(0);
        __syncthreads();
    }

    // ---- epilogue
    const size_t orow0 = (size_t)q0 * E_;
    #pragma unroll
    for (int r = 0; r < 8; ++r) {
        const float inv = 1.0f / lstate[r];
        #pragma unroll
        for (int j = 0; j < 8; ++j)
            out[orow0 + (size_t)(r + 8 * hi) * E_ + j * 16 + lr] = O[j][r] * inv;
    }
}

// ---------------------------------------------------------------------------
extern "C" void kernel_launch(void* const* d_in, const int* in_sizes, int n_in,
                              void* d_out, int out_size, void* d_ws, size_t ws_size,
                              hipStream_t stream)
{
    const float* X  = (const float*)d_in[0];
    // d_in[1] (context), d_in[8] (Wc), d_in[9] (bc): bias is constant along the
    // softmax (key) axis, hence mathematically a no-op -> unused.
    const float* Wq = (const float*)d_in[2];
    const float* bq = (const float*)d_in[3];
    const float* Wk = (const float*)d_in[4];
    const float* bk = (const float*)d_in[5];
    const float* Wv = (const float*)d_in[6];
    const float* bv = (const float*)d_in[7];
    float* out = (float*)d_out;

    _Float16* Qh = (_Float16*)d_ws;                       // [B*N][E] f16
    _Float16* Kh = Qh + (size_t)B_ * N_ * E_;             // [B*N][E] f16
    _Float16* Vt = Kh + (size_t)B_ * N_ * E_;             // [B][E][N] f16

    qkv_proj_kernel<<<dim3(B_ * N_ / 16, 3), 256, 0, stream>>>(
        X, Wq, bq, Wk, bk, Wv, bv, Qh, Kh, Vt);
    attn_kernel<<<dim3(B_ * N_ / 128), 256, 0, stream>>>(Qh, Kh, Vt, out);
}
